// GeometricLongShortRNA_19567871000999
// MI455X (gfx1250) — compile-verified, matrix operand order checked by reading, not run on hardware
//
#include <hip/hip_runtime.h>
#include <hip/hip_bf16.h>

// ---------------------------------------------------------------------------
// Types
// ---------------------------------------------------------------------------
typedef __bf16 bf16;
typedef __attribute__((ext_vector_type(16))) __bf16 v16bf;
typedef __attribute__((ext_vector_type(8)))  float  v8f;

__device__ __forceinline__ float siluf(float x) { return x / (1.f + __expf(-x)); }
__device__ __forceinline__ float sigmf(float x) { return 1.f / (1.f + __expf(-x)); }

// ---------------------------------------------------------------------------
// WMMA helper: one wave computes a 16x16 tile of  Sl[16][SOW] (+bias),
// A is a bf16 LDS tile [16][KP], Wt is bf16 global [SO][KP] (N-major,
// i.e. pre-transposed + zero-padded weights).  K loop in steps of 32.
// bf16 operand layout per CDNA5 ISA 7.12.2 (16-bit A 16x32):
//   lanes 0-15 : M=lane,    K = k0+0..7  and k0+16..23
//   lanes 16-31: M=lane-16, K = k0+8..15 and k0+24..31
// B uses the mirrored layout on the N index (weights stored N-major).
// ---------------------------------------------------------------------------
__device__ __forceinline__ void wmma_gemm_tile(
    const bf16* __restrict__ A, int KP,
    const bf16* __restrict__ Wt, const float* __restrict__ bias,
    float* __restrict__ Sl, int SOW, int ncol0, int tid)
{
  const int lane = tid & 31;
  const int mrow = lane & 15;
  const int hi   = lane >> 4;
  v8f C = {0.f,0.f,0.f,0.f,0.f,0.f,0.f,0.f};
  for (int k0 = 0; k0 < KP; k0 += 32) {
    v16bf a, b;
    const bf16* ap = A  + (size_t)mrow * KP + k0 + hi * 8;
    const bf16* bp = Wt + (size_t)(ncol0 + mrow) * KP + k0 + hi * 8;
    __builtin_prefetch(bp + 32, 0, 3);
#pragma unroll
    for (int i = 0; i < 8; ++i) {
      a[i]     = ap[i];
      a[8 + i] = ap[16 + i];
      b[i]     = bp[i];
      b[8 + i] = bp[16 + i];
    }
    C = __builtin_amdgcn_wmma_f32_16x16x32_bf16(
        /*neg_a=*/false, a, /*neg_b=*/false, b,
        /*c_mod=*/(short)0, C, /*reuse_a=*/false, /*reuse_b=*/false);
  }
#pragma unroll
  for (int r = 0; r < 8; ++r)
    Sl[(size_t)(hi * 8 + r) * SOW + ncol0 + mrow] = C[r] + bias[ncol0 + mrow];
}

// ---------------------------------------------------------------------------
// Weight pre-transpose: out[n][k] = (k<K ? w[k][n] : 0) as bf16.
// ---------------------------------------------------------------------------
__global__ void transpose_pad_kernel(const float* __restrict__ w,
                                     bf16* __restrict__ out,
                                     int K, int N, int KPAD)
{
  int idx = blockIdx.x * blockDim.x + threadIdx.x;
  if (idx >= N * KPAD) return;
  int n = idx / KPAD, k = idx % KPAD;
  out[idx] = (bf16)((k < K) ? w[(size_t)k * N + n] : 0.f);
}

// ---------------------------------------------------------------------------
// LayerNorm over scalars + vector RMS-norm per node (GVP layer_norm).
// blockDim = 128, one block per node.
// ---------------------------------------------------------------------------
__global__ __launch_bounds__(128) void ln_node_kernel(
    const float* __restrict__ s_in, const float* __restrict__ v_in,
    const float* __restrict__ g, const float* __restrict__ b,
    float* __restrict__ s_out, float* __restrict__ v_out, int S, int V)
{
  __shared__ float red[128];
  __shared__ float stat[2];
  int n = blockIdx.x, t = threadIdx.x;
  float x = (t < S) ? s_in[(size_t)n * S + t] : 0.f;
  red[t] = x; __syncthreads();
  for (int o = 64; o > 0; o >>= 1) { if (t < o) red[t] += red[t + o]; __syncthreads(); }
  if (t == 0) stat[0] = red[0] / (float)S;
  __syncthreads();
  float mu = stat[0];
  float d = (t < S) ? (x - mu) : 0.f;
  red[t] = d * d; __syncthreads();
  for (int o = 64; o > 0; o >>= 1) { if (t < o) red[t] += red[t + o]; __syncthreads(); }
  if (t == 0) stat[1] = rsqrtf(red[0] / (float)S + 1e-5f);
  __syncthreads();
  if (t < S) s_out[(size_t)n * S + t] = d * stat[1] * g[t] + b[t];
  // vectors: vn = sqrt(mean_v max(|v|^2, eps)) ; v /= vn
  float sq = 0.f;
  if (t < V) {
    const float* vp = v_in + ((size_t)n * V + t) * 3;
    sq = fmaxf(vp[0]*vp[0] + vp[1]*vp[1] + vp[2]*vp[2], 1e-8f);
  }
  __syncthreads();
  red[t] = (t < V) ? sq : 0.f; __syncthreads();
  for (int o = 64; o > 0; o >>= 1) { if (t < o) red[t] += red[t + o]; __syncthreads(); }
  if (t == 0) stat[0] = rsqrtf(red[0] / (float)V);
  __syncthreads();
  float vs = stat[0];
  for (int i = t; i < V * 3; i += 128)
    v_out[(size_t)n * V * 3 + i] = v_in[(size_t)n * V * 3 + i] * vs;
}

// ---------------------------------------------------------------------------
// Generic per-node GVP (vector_gate=true), WMMA scalar path + WMMA gate path.
// 16 nodes / block, 8 waves; wave w handles output column tiles w, w+8, ...
// ---------------------------------------------------------------------------
template <int KPAD, int SOMAX>
__global__ __launch_bounds__(256) void gvp_node_kernel(
    const float* __restrict__ s_in, const float* __restrict__ v_in,
    int M, int SI, int VI, int H, int SO, int VO,
    const float* __restrict__ wh,  const float* __restrict__ wsb,
    const bf16*  __restrict__ wsw, const float* __restrict__ wsvb,
    const bf16*  __restrict__ wsvt, const float* __restrict__ wv,
    int do_silu,
    float* __restrict__ s_out, float* __restrict__ v_out)
{
  __shared__ bf16  As[16][KPAD];
  __shared__ float Sl[16][SOMAX];
  __shared__ bf16  Ag[16][SOMAX];
  __shared__ float G[16][32];
  __shared__ float Vh[16][33][3];
  int tid = threadIdx.x;
  int m0 = blockIdx.x * 16;
  // A tile: scalar features, pad zero (vn part overwritten below)
  for (int idx = tid; idx < 16 * KPAD; idx += 256) {
    int e = idx / KPAD, c = idx % KPAD;
    float val = 0.f;
    if (c < SI && m0 + e < M) val = s_in[(size_t)(m0 + e) * SI + c];
    As[e][c] = (bf16)val;
  }
  __syncthreads();
  // vh[e][h][c] = sum_v v[e][v][c] * wh[v][h]
  for (int idx = tid; idx < 16 * H * 3; idx += 256) {
    int e = idx / (H * 3), r = idx % (H * 3), h = r / 3, c = r % 3;
    float acc = 0.f;
    if (m0 + e < M)
      for (int v = 0; v < VI; ++v)
        acc += v_in[((size_t)(m0 + e) * VI + v) * 3 + c] * wh[v * H + h];
    Vh[e][h][c] = acc;
  }
  __syncthreads();
  for (int idx = tid; idx < 16 * H; idx += 256) {
    int e = idx / H, h = idx % H;
    float n2 = Vh[e][h][0]*Vh[e][h][0] + Vh[e][h][1]*Vh[e][h][1] + Vh[e][h][2]*Vh[e][h][2];
    As[e][SI + h] = (bf16)sqrtf(fmaxf(n2, 1e-8f));
  }
  __syncthreads();
  // GEMM: s_lin = A @ Wsw + b
  int wave = tid >> 5;
  for (int nt = wave; nt * 16 < SO; nt += 8)
    wmma_gemm_tile(&As[0][0], KPAD, wsw, wsb, &Sl[0][0], SOMAX, nt * 16, tid);
  __syncthreads();
  // gate GEMM input: bf16 copy of pre-activation s_lin
  for (int idx = tid; idx < 16 * SO; idx += 256) {
    int e = idx / SO, c = idx % SO;
    Ag[e][c] = (bf16)Sl[e][c];
  }
  __syncthreads();
  // gate = s_lin @ wsv_w + b  (16 x VO tile, K = SO) via WMMA
  for (int nt = wave; nt * 16 < VO; nt += 8)
    wmma_gemm_tile(&Ag[0][0], SO, wsvt, wsvb, &G[0][0], 32, nt * 16, tid);
  __syncthreads();
  // vout * sigmoid(gate)
  for (int idx = tid; idx < 16 * VO; idx += 256) {
    int e = idx / VO, vv = idx % VO;
    if (m0 + e >= M) continue;
    float gg = sigmf(G[e][vv]);
    for (int c = 0; c < 3; ++c) {
      float acc = 0.f;
      for (int h = 0; h < H; ++h) acc += Vh[e][h][c] * wv[h * VO + vv];
      v_out[((size_t)(m0 + e) * VO + vv) * 3 + c] = acc * gg;
    }
  }
  for (int idx = tid; idx < 16 * SO; idx += 256) {
    int e = idx / SO, c = idx % SO;
    if (m0 + e >= M) continue;
    float x = Sl[e][c];
    s_out[(size_t)(m0 + e) * SO + c] = do_silu ? siluf(x) : x;
  }
}

// ---------------------------------------------------------------------------
// Fused edge-message kernel: 3 chained GVPs per edge, 16 edges / block.
// Scalar GEMMs and the three sigmoid-gate GEMMs all on WMMA.
// ---------------------------------------------------------------------------
struct ConvW {
  const float *wh0, *wsb0, *wsvb0, *wv0; const bf16 *wsw0, *wsvt0;
  const float *wh1, *wsb1, *wsvb1, *wv1; const bf16 *wsw1, *wsvt1;
  const float *wh2, *wsb2, *wsvb2, *wv2; const bf16 *wsw2, *wsvt2;
};

__global__ __launch_bounds__(256) void edge_message_kernel(
    const float* __restrict__ sn,    const float* __restrict__ vnb,
    const float* __restrict__ arn_s, const float* __restrict__ arn_v,
    const float* __restrict__ es, int SIE,
    const float* __restrict__ ev,
    const int* __restrict__ src, const int* __restrict__ dst,
    ConvW W, float* __restrict__ acc_s, float* __restrict__ acc_v, int E)
{
  __shared__ bf16  As0[16][352];
  __shared__ bf16  As1[16][160];
  __shared__ bf16  Ag[16][128];
  __shared__ float Sl[16][128];
  __shared__ float G[16][16];
  __shared__ float Vc[16][33][3];
  __shared__ float Vh[16][33][3];
  __shared__ float Vt[16][16][3];
  __shared__ float Vh2[16][16][3];
  __shared__ int   Se[16], De[16];

  const int tid = threadIdx.x;
  const int wave = tid >> 5;
  const int e0 = blockIdx.x * 16;
  if (tid < 16) {
    int e = e0 + tid; if (e >= E) e = E - 1;
    Se[tid] = src[e]; De[tid] = dst[e];
  }
  __syncthreads();

  // ---- build A0 scalar tile: [s_j | es | s_i | vn(33) | pad] -> 352 cols
  for (int idx = tid; idx < 16 * 352; idx += 256) {
    int el = idx / 352, c = idx % 352;
    int s_ = Se[el], d_ = De[el];
    bool m = s_ < d_;                       // decoder AR select (enc: sn==arn)
    int eg = e0 + el; if (eg >= E) eg = E - 1;
    float val = 0.f;
    if (c < 128)             { const float* pS = m ? sn : arn_s; val = pS[(size_t)s_ * 128 + c]; }
    else if (c < 128 + SIE)  { val = es[(size_t)eg * SIE + (c - 128)]; }
    else if (c < 256 + SIE)  { const float* pS = m ? sn : arn_s; val = pS[(size_t)d_ * 128 + (c - 128 - SIE)]; }
    As0[el][c] = (bf16)val;                 // vn cols overwritten later
  }
  // ---- vector concat: [v_j(16) | ev(1) | v_i(16)]
  for (int idx = tid; idx < 16 * 33 * 3; idx += 256) {
    int el = idx / 99, r = idx % 99, vv = r / 3, c = r % 3;
    int s_ = Se[el], d_ = De[el];
    bool m = s_ < d_;
    int eg = e0 + el; if (eg >= E) eg = E - 1;
    float val;
    if (vv < 16)       { const float* pV = m ? vnb : arn_v; val = pV[((size_t)s_ * 16 + vv) * 3 + c]; }
    else if (vv == 16) { val = ev[(size_t)eg * 3 + c]; }
    else               { const float* pV = m ? vnb : arn_v; val = pV[((size_t)d_ * 16 + (vv - 17)) * 3 + c]; }
    Vc[el][vv][c] = val;
  }
  __syncthreads();
  // ---- vh0 = Vc @ wh0 (33x33)
  for (int idx = tid; idx < 16 * 33 * 3; idx += 256) {
    int el = idx / 99, r = idx % 99, h = r / 3, c = r % 3;
    float acc = 0.f;
    for (int v = 0; v < 33; ++v) acc += Vc[el][v][c] * W.wh0[v * 33 + h];
    Vh[el][h][c] = acc;
  }
  __syncthreads();
  for (int idx = tid; idx < 16 * 33; idx += 256) {
    int el = idx / 33, h = idx % 33;
    float n2 = Vh[el][h][0]*Vh[el][h][0] + Vh[el][h][1]*Vh[el][h][1] + Vh[el][h][2]*Vh[el][h][2];
    As0[el][256 + SIE + h] = (bf16)sqrtf(fmaxf(n2, 1e-8f));
  }
  __syncthreads();
  // ---- GVP0 scalar GEMM (K=352 padded)
  wmma_gemm_tile(&As0[0][0], 352, W.wsw0, W.wsb0, &Sl[0][0], 128, wave * 16, tid);
  __syncthreads();
  // ---- bf16 copies: gate input (pre-act) + next GEMM A (silu)
  for (int idx = tid; idx < 16 * 128; idx += 256) {
    int el = idx / 128, c = idx % 128;
    float x = Sl[el][c];
    Ag[el][c]  = (bf16)x;
    As1[el][c] = (bf16)siluf(x);
  }
  for (int idx = tid; idx < 16 * 16; idx += 256)
    As1[idx / 16][144 + (idx & 15)] = (bf16)0.f;   // K pad 144..159
  __syncthreads();
  // ---- gate0 GEMM (16x16, K=128) on wave 0
  if (wave == 0)
    wmma_gemm_tile(&Ag[0][0], 128, W.wsvt0, W.wsvb0, &G[0][0], 16, 0, tid);
  __syncthreads();
  // ---- v1 = sigmoid(gate0) * (vh0 @ wv0)
  for (int idx = tid; idx < 16 * 16; idx += 256) {
    int el = idx / 16, vv = idx % 16;
    float gg = sigmf(G[el][vv]);
    for (int c = 0; c < 3; ++c) {
      float acc = 0.f;
      for (int h = 0; h < 33; ++h) acc += Vh[el][h][c] * W.wv0[h * 16 + vv];
      Vt[el][vv][c] = acc * gg;
    }
  }
  __syncthreads();
  // ---- vh2 = v1 @ wh1 (16x16)
  for (int idx = tid; idx < 16 * 16 * 3; idx += 256) {
    int el = idx / 48, r = idx % 48, h = r / 3, c = r % 3;
    float acc = 0.f;
    for (int v = 0; v < 16; ++v) acc += Vt[el][v][c] * W.wh1[v * 16 + h];
    Vh2[el][h][c] = acc;
  }
  __syncthreads();
  for (int idx = tid; idx < 16 * 16; idx += 256) {
    int el = idx / 16, h = idx % 16;
    float n2 = Vh2[el][h][0]*Vh2[el][h][0] + Vh2[el][h][1]*Vh2[el][h][1] + Vh2[el][h][2]*Vh2[el][h][2];
    As1[el][128 + h] = (bf16)sqrtf(fmaxf(n2, 1e-8f));
  }
  __syncthreads();
  // ---- GVP1 scalar GEMM (K=160)
  wmma_gemm_tile(&As1[0][0], 160, W.wsw1, W.wsb1, &Sl[0][0], 128, wave * 16, tid);
  __syncthreads();
  for (int idx = tid; idx < 16 * 128; idx += 256) {
    int el = idx / 128, c = idx % 128;
    float x = Sl[el][c];
    Ag[el][c]  = (bf16)x;
    As1[el][c] = (bf16)siluf(x);
  }
  __syncthreads();
  // ---- gate1 GEMM
  if (wave == 0)
    wmma_gemm_tile(&Ag[0][0], 128, W.wsvt1, W.wsvb1, &G[0][0], 16, 0, tid);
  __syncthreads();
  // ---- v2 = sigmoid(gate1) * (vh2 @ wv1)
  for (int idx = tid; idx < 16 * 16; idx += 256) {
    int el = idx / 16, vv = idx % 16;
    float gg = sigmf(G[el][vv]);
    for (int c = 0; c < 3; ++c) {
      float acc = 0.f;
      for (int h = 0; h < 16; ++h) acc += Vh2[el][h][c] * W.wv1[h * 16 + vv];
      Vt[el][vv][c] = acc * gg;             // v2 (v1 dead)
    }
  }
  __syncthreads();
  // ---- vh3 = v2 @ wh2
  for (int idx = tid; idx < 16 * 16 * 3; idx += 256) {
    int el = idx / 48, r = idx % 48, h = r / 3, c = r % 3;
    float acc = 0.f;
    for (int v = 0; v < 16; ++v) acc += Vt[el][v][c] * W.wh2[v * 16 + h];
    Vh2[el][h][c] = acc;
  }
  __syncthreads();
  for (int idx = tid; idx < 16 * 16; idx += 256) {
    int el = idx / 16, h = idx % 16;
    float n2 = Vh2[el][h][0]*Vh2[el][h][0] + Vh2[el][h][1]*Vh2[el][h][1] + Vh2[el][h][2]*Vh2[el][h][2];
    As1[el][128 + h] = (bf16)sqrtf(fmaxf(n2, 1e-8f));
  }
  __syncthreads();
  // ---- GVP2 scalar GEMM (no scalar activation)
  wmma_gemm_tile(&As1[0][0], 160, W.wsw2, W.wsb2, &Sl[0][0], 128, wave * 16, tid);
  __syncthreads();
  for (int idx = tid; idx < 16 * 128; idx += 256)
    Ag[idx / 128][idx & 127] = (bf16)Sl[idx / 128][idx & 127];
  __syncthreads();
  // ---- gate2 GEMM
  if (wave == 0)
    wmma_gemm_tile(&Ag[0][0], 128, W.wsvt2, W.wsvb2, &G[0][0], 16, 0, tid);
  __syncthreads();
  // ---- v3 = sigmoid(gate2) * (vh3 @ wv2); scatter msg with atomics
  for (int idx = tid; idx < 16 * 16; idx += 256) {
    int el = idx / 16, vv = idx % 16;
    if (e0 + el >= E) continue;
    float gg = sigmf(G[el][vv]);
    int d_ = De[el];
    for (int c = 0; c < 3; ++c) {
      float acc = 0.f;
      for (int h = 0; h < 16; ++h) acc += Vh2[el][h][c] * W.wv2[h * 16 + vv];
      atomicAdd(&acc_v[((size_t)d_ * 16 + vv) * 3 + c], acc * gg);
    }
  }
  for (int idx = tid; idx < 16 * 128; idx += 256) {
    int el = idx / 128, c = idx % 128;
    if (e0 + el >= E) continue;
    atomicAdd(&acc_s[(size_t)De[el] * 128 + c], Sl[el][c]);
  }
}

// ---------------------------------------------------------------------------
// Edge embedding: fused We_ln + We GVP (tiny dims: 33->32 scalars, 1->1 vec)
// ---------------------------------------------------------------------------
__global__ void edge_embed_kernel(
    const float* __restrict__ edge_s, const float* __restrict__ edge_v,
    const float* __restrict__ ln_g, const float* __restrict__ ln_b,
    const float* __restrict__ wh, const float* __restrict__ wsb,
    const float* __restrict__ wsw, const float* __restrict__ wsvb,
    const float* __restrict__ wsvw, const float* __restrict__ wv,
    float* __restrict__ es_out, float* __restrict__ ev_out, int E)
{
  int e = blockIdx.x * blockDim.x + threadIdx.x;
  if (e >= E) return;
  const float* sp = edge_s + (size_t)e * 32;
  float s[32];
  float mu = 0.f;
  for (int i = 0; i < 32; ++i) mu += sp[i];
  mu *= (1.f / 32.f);
  float var = 0.f;
  for (int i = 0; i < 32; ++i) { float d = sp[i] - mu; var += d * d; }
  float inv = rsqrtf(var * (1.f / 32.f) + 1e-5f);
  for (int i = 0; i < 32; ++i) s[i] = (sp[i] - mu) * inv * ln_g[i] + ln_b[i];
  const float* vp = edge_v + (size_t)e * 3;
  float n2 = fmaxf(vp[0]*vp[0] + vp[1]*vp[1] + vp[2]*vp[2], 1e-8f);
  float vs = rsqrtf(n2);                 // ln: v / sqrt(mean(|v|^2)) with 1 vector
  float vh[3];
  for (int c = 0; c < 3; ++c) vh[c] = vp[c] * vs * wh[0];
  float vhn = sqrtf(fmaxf(vh[0]*vh[0] + vh[1]*vh[1] + vh[2]*vh[2], 1e-8f));
  float so[32];
  for (int o = 0; o < 32; ++o) {
    float acc = wsb[o];
    for (int k = 0; k < 32; ++k) acc += s[k] * wsw[k * 32 + o];
    acc += vhn * wsw[32 * 32 + o];
    so[o] = acc;
  }
  float gg = wsvb[0];
  for (int o = 0; o < 32; ++o) gg += so[o] * wsvw[o];
  gg = sigmf(gg);
  for (int o = 0; o < 32; ++o) es_out[(size_t)e * 32 + o] = so[o];
  for (int c = 0; c < 3; ++c)  ev_out[(size_t)e * 3 + c] = vh[c] * wv[0] * gg;
}

// ---------------------------------------------------------------------------
// Misc small kernels
// ---------------------------------------------------------------------------
__global__ void count_kernel(const int* __restrict__ dst, float* __restrict__ cnt, int E)
{
  int e = blockIdx.x * blockDim.x + threadIdx.x;
  if (e < E) atomicAdd(&cnt[dst[e]], 1.f);
}

__global__ void segmean_residual_kernel(
    float* __restrict__ cur_s, float* __restrict__ cur_v,
    const float* __restrict__ acc_s, const float* __restrict__ acc_v,
    const float* __restrict__ cnt, int N)
{
  int idx = blockIdx.x * blockDim.x + threadIdx.x;
  int tot = N * (128 + 48);
  if (idx >= tot) return;
  if (idx < N * 128) {
    int n = idx / 128;
    cur_s[idx] += acc_s[idx] / fmaxf(cnt[n], 1.f);
  } else {
    int j = idx - N * 128;
    int n = j / 48;
    cur_v[j] += acc_v[j] / fmaxf(cnt[n], 1.f);
  }
}

__global__ void residual_add_kernel(
    float* __restrict__ cur_s, float* __restrict__ cur_v,
    const float* __restrict__ ts, const float* __restrict__ tv, int N)
{
  int idx = blockIdx.x * blockDim.x + threadIdx.x;
  int tot = N * (128 + 48);
  if (idx >= tot) return;
  if (idx < N * 128) cur_s[idx] += ts[idx];
  else { int j = idx - N * 128; cur_v[j] += tv[j]; }
}

__global__ void esdec_kernel(
    const float* __restrict__ es, const int* __restrict__ src,
    const int* __restrict__ dst, const int* __restrict__ seq,
    const float* __restrict__ W_s, float* __restrict__ es_dec, int E)
{
  int idx = blockIdx.x * blockDim.x + threadIdx.x;
  if (idx >= E * 36) return;
  int e = idx / 36, c = idx % 36;
  if (c < 32) es_dec[idx] = es[(size_t)e * 32 + c];
  else {
    int j = c - 32;
    int s_ = src[e];
    es_dec[idx] = (s_ < dst[e]) ? W_s[seq[s_] * 4 + j] : 0.f;
  }
}

__global__ void wout_kernel(
    const float* __restrict__ s, const float* __restrict__ v,
    const float* __restrict__ wh, const float* __restrict__ wsb,
    const float* __restrict__ wsw, float* __restrict__ out, int N)
{
  int n = blockIdx.x * blockDim.x + threadIdx.x;
  if (n >= N) return;
  float vn[16];
  for (int h = 0; h < 16; ++h) {
    float a0 = 0.f, a1 = 0.f, a2 = 0.f;
    for (int vv = 0; vv < 16; ++vv) {
      float w = wh[vv * 16 + h];
      const float* vp = v + ((size_t)n * 16 + vv) * 3;
      a0 += vp[0] * w; a1 += vp[1] * w; a2 += vp[2] * w;
    }
    vn[h] = sqrtf(fmaxf(a0*a0 + a1*a1 + a2*a2, 1e-8f));
  }
  for (int o = 0; o < 4; ++o) {
    float acc = wsb[o];
    for (int k = 0; k < 128; ++k) acc += s[(size_t)n * 128 + k] * wsw[k * 4 + o];
    for (int h = 0; h < 16; ++h)  acc += vn[h] * wsw[(128 + h) * 4 + o];
    out[(size_t)n * 4 + o] = acc;
  }
}

// ---------------------------------------------------------------------------
// Host orchestration
// ---------------------------------------------------------------------------
struct GvpP   { const float *wh, *ws_b, *ws_w, *wsv_b, *wsv_w, *wv; };
struct LayerP { GvpP c0, c1, c2, f0, f1; const float *n0b, *n0g, *n1b, *n1g; };
struct LayerW { const bf16 *c0, *c1, *c2, *f0, *f1;
                const bf16 *g0, *g1, *g2, *gf0, *gf1; };

extern "C" void kernel_launch(void* const* d_in, const int* in_sizes, int n_in,
                              void* d_out, int out_size, void* d_ws, size_t ws_size,
                              hipStream_t stream)
{
  const float* node_s = (const float*)d_in[0];   // [N,1,64]
  const float* node_v = (const float*)d_in[1];   // [N,1,4,3]
  const float* edge_s = (const float*)d_in[2];   // [E,1,32]
  const float* edge_v = (const float*)d_in[3];   // [E,1,1,3]
  const int*   eidx   = (const int*)d_in[4];     // [2,E]
  const int*   seq    = (const int*)d_in[5];     // [N]
  // d_in[6] = mask_confs (all ones, unused)

  const int N = in_sizes[0] / 64;
  const int E = in_sizes[2] / 32;
  const int* src = eidx;
  const int* dst = eidx + E;

  // ---- param leaves (JAX pytree: dict keys sorted) -----------------------
  int p = 7;
  auto F = [&](int i) { return (const float*)d_in[i]; };
  auto loadGvp6 = [&](int& q) { GvpP g; g.wh = F(q++); g.ws_b = F(q++); g.ws_w = F(q++);
                                g.wsv_b = F(q++); g.wsv_w = F(q++); g.wv = F(q++); return g; };
  const float* W_s = F(p++);                      // (4,4)
  GvpP We = loadGvp6(p);
  const float* We_ln_b = F(p++); const float* We_ln_g = F(p++);
  const float* Wout_wh = F(p++); const float* Wout_b = F(p++); const float* Wout_w = F(p++);
  GvpP Wv = loadGvp6(p);
  const float* Wv_ln_b = F(p++); const float* Wv_ln_g = F(p++);
  auto loadLayer = [&](int& q) {
    LayerP L; L.c0 = loadGvp6(q); L.c1 = loadGvp6(q); L.c2 = loadGvp6(q);
    L.f0 = loadGvp6(q); L.f1 = loadGvp6(q);
    L.n0b = F(q++); L.n0g = F(q++); L.n1b = F(q++); L.n1g = F(q++); return L; };
  LayerP decP[3], encP[3];
  for (int l = 0; l < 3; ++l) decP[l] = loadLayer(p);
  for (int l = 0; l < 3; ++l) encP[l] = loadLayer(p);

  // ---- workspace ----------------------------------------------------------
  size_t off = 0;
  auto alloc = [&](size_t bytes) -> char* {
    char* r = (char*)d_ws + off; off += (bytes + 255) & ~(size_t)255; return r; };
  float* sn     = (float*)alloc((size_t)N * 128 * 4);
  float* vnb    = (float*)alloc((size_t)N * 48 * 4);
  float* arn_s  = (float*)alloc((size_t)N * 128 * 4);
  float* arn_v  = (float*)alloc((size_t)N * 48 * 4);
  float* enc_s  = (float*)alloc((size_t)N * 128 * 4);
  float* enc_v  = (float*)alloc((size_t)N * 48 * 4);
  float* cur_s  = (float*)alloc((size_t)N * 128 * 4);
  float* cur_v  = (float*)alloc((size_t)N * 48 * 4);
  float* acc_s  = (float*)alloc((size_t)N * 128 * 4);
  float* acc_v  = (float*)alloc((size_t)N * 48 * 4);
  float* tmp_s  = (float*)alloc((size_t)N * 128 * 4);
  float* tmp_v  = (float*)alloc((size_t)N * 48 * 4);
  float* fs     = (float*)alloc((size_t)N * 512 * 4);
  float* fv     = (float*)alloc((size_t)N * 96 * 4);
  float* cnt    = (float*)alloc((size_t)N * 4);
  float* es     = (float*)alloc((size_t)E * 32 * 4);
  float* evb    = (float*)alloc((size_t)E * 3 * 4);
  float* es_dec = (float*)alloc((size_t)E * 36 * 4);

  // ---- bf16 weight conversion (transpose + pad to K multiple of 32) ------
  auto convw = [&](const float* w, int K, int Nc, int KPAD) -> const bf16* {
    bf16* o = (bf16*)alloc((size_t)Nc * KPAD * 2);
    int tot = Nc * KPAD;
    transpose_pad_kernel<<<(tot + 255) / 256, 256, 0, stream>>>(w, o, K, Nc, KPAD);
    return o;
  };
  const bf16* WvT  = convw(Wv.ws_w, 80, 128, 96);
  const bf16* WvG  = convw(Wv.wsv_w, 128, 16, 128);
  LayerW encW[3], decW[3];
  for (int l = 0; l < 3; ++l) {
    encW[l].c0  = convw(encP[l].c0.ws_w, 321, 128, 352);
    encW[l].c1  = convw(encP[l].c1.ws_w, 144, 128, 160);
    encW[l].c2  = convw(encP[l].c2.ws_w, 144, 128, 160);
    encW[l].f0  = convw(encP[l].f0.ws_w, 160, 512, 160);
    encW[l].f1  = convw(encP[l].f1.ws_w, 544, 128, 544);
    encW[l].g0  = convw(encP[l].c0.wsv_w, 128, 16, 128);
    encW[l].g1  = convw(encP[l].c1.wsv_w, 128, 16, 128);
    encW[l].g2  = convw(encP[l].c2.wsv_w, 128, 16, 128);
    encW[l].gf0 = convw(encP[l].f0.wsv_w, 512, 32, 512);
    encW[l].gf1 = convw(encP[l].f1.wsv_w, 128, 16, 128);
    decW[l].c0  = convw(decP[l].c0.ws_w, 325, 128, 352);
    decW[l].c1  = convw(decP[l].c1.ws_w, 144, 128, 160);
    decW[l].c2  = convw(decP[l].c2.ws_w, 144, 128, 160);
    decW[l].f0  = convw(decP[l].f0.ws_w, 160, 512, 160);
    decW[l].f1  = convw(decP[l].f1.ws_w, 544, 128, 544);
    decW[l].g0  = convw(decP[l].c0.wsv_w, 128, 16, 128);
    decW[l].g1  = convw(decP[l].c1.wsv_w, 128, 16, 128);
    decW[l].g2  = convw(decP[l].c2.wsv_w, 128, 16, 128);
    decW[l].gf0 = convw(decP[l].f0.wsv_w, 512, 32, 512);
    decW[l].gf1 = convw(decP[l].f1.wsv_w, 128, 16, 128);
  }
  auto mkConv = [&](const LayerP& L, const LayerW& LW) {
    ConvW c;
    c.wh0 = L.c0.wh; c.wsb0 = L.c0.ws_b; c.wsvb0 = L.c0.wsv_b; c.wv0 = L.c0.wv;
    c.wsw0 = LW.c0; c.wsvt0 = LW.g0;
    c.wh1 = L.c1.wh; c.wsb1 = L.c1.ws_b; c.wsvb1 = L.c1.wsv_b; c.wv1 = L.c1.wv;
    c.wsw1 = LW.c1; c.wsvt1 = LW.g1;
    c.wh2 = L.c2.wh; c.wsb2 = L.c2.ws_b; c.wsvb2 = L.c2.wsv_b; c.wv2 = L.c2.wv;
    c.wsw2 = LW.c2; c.wsvt2 = LW.g2;
    return c;
  };

  const int NB16 = (N + 15) / 16;
  const int EB16 = (E + 15) / 16;

  // ---- degree counts (constant graph) -------------------------------------
  hipMemsetAsync(cnt, 0, (size_t)N * 4, stream);
  count_kernel<<<(E + 255) / 256, 256, 0, stream>>>(dst, cnt, E);

  // ---- input embeddings ----------------------------------------------------
  ln_node_kernel<<<N, 128, 0, stream>>>(node_s, node_v, Wv_ln_g, Wv_ln_b, fs, fv, 64, 4);
  gvp_node_kernel<96, 128><<<NB16, 256, 0, stream>>>(
      fs, fv, N, 64, 4, 16, 128, 16,
      Wv.wh, Wv.ws_b, WvT, Wv.wsv_b, WvG, Wv.wv, 0, cur_s, cur_v);
  edge_embed_kernel<<<(E + 255) / 256, 256, 0, stream>>>(
      edge_s, edge_v, We_ln_g, We_ln_b,
      We.wh, We.ws_b, We.ws_w, We.wsv_b, We.wsv_w, We.wv, es, evb, E);

  // ---- encoder -------------------------------------------------------------
  for (int l = 0; l < 3; ++l) {
    const LayerP& L = encP[l];
    ln_node_kernel<<<N, 128, 0, stream>>>(cur_s, cur_v, L.n0g, L.n0b, sn, vnb, 128, 16);
    hipMemsetAsync(acc_s, 0, (size_t)N * 128 * 4, stream);
    hipMemsetAsync(acc_v, 0, (size_t)N * 48 * 4, stream);
    ConvW cw = mkConv(L, encW[l]);
    edge_message_kernel<<<EB16, 256, 0, stream>>>(
        sn, vnb, sn, vnb, es, 32, evb, src, dst, cw, acc_s, acc_v, E);
    segmean_residual_kernel<<<(N * 176 + 255) / 256, 256, 0, stream>>>(
        cur_s, cur_v, acc_s, acc_v, cnt, N);
    ln_node_kernel<<<N, 128, 0, stream>>>(cur_s, cur_v, L.n1g, L.n1b, sn, vnb, 128, 16);
    gvp_node_kernel<160, 512><<<NB16, 256, 0, stream>>>(
        sn, vnb, N, 128, 16, 32, 512, 32,
        L.f0.wh, L.f0.ws_b, encW[l].f0, L.f0.wsv_b, encW[l].gf0, L.f0.wv, 1, fs, fv);
    gvp_node_kernel<544, 128><<<NB16, 256, 0, stream>>>(
        fs, fv, N, 512, 32, 32, 128, 16,
        L.f1.wh, L.f1.ws_b, encW[l].f1, L.f1.wsv_b, encW[l].gf1, L.f1.wv, 0, tmp_s, tmp_v);
    residual_add_kernel<<<(N * 176 + 255) / 256, 256, 0, stream>>>(cur_s, cur_v, tmp_s, tmp_v, N);
  }

  // ---- save encoder output (autoregressive context) -----------------------
  hipMemcpyAsync(enc_s, cur_s, (size_t)N * 128 * 4, hipMemcpyDeviceToDevice, stream);
  hipMemcpyAsync(enc_v, cur_v, (size_t)N * 48 * 4, hipMemcpyDeviceToDevice, stream);

  // ---- decoder edge scalars: [es | h_S] ------------------------------------
  esdec_kernel<<<(E * 36 + 255) / 256, 256, 0, stream>>>(es, src, dst, seq, W_s, es_dec, E);

  // ---- decoder -------------------------------------------------------------
  for (int l = 0; l < 3; ++l) {
    const LayerP& L = decP[l];
    ln_node_kernel<<<N, 128, 0, stream>>>(cur_s, cur_v, L.n0g, L.n0b, sn, vnb, 128, 16);
    ln_node_kernel<<<N, 128, 0, stream>>>(enc_s, enc_v, L.n0g, L.n0b, arn_s, arn_v, 128, 16);
    hipMemsetAsync(acc_s, 0, (size_t)N * 128 * 4, stream);
    hipMemsetAsync(acc_v, 0, (size_t)N * 48 * 4, stream);
    ConvW cw = mkConv(L, decW[l]);
    edge_message_kernel<<<EB16, 256, 0, stream>>>(
        sn, vnb, arn_s, arn_v, es_dec, 36, evb, src, dst, cw, acc_s, acc_v, E);
    segmean_residual_kernel<<<(N * 176 + 255) / 256, 256, 0, stream>>>(
        cur_s, cur_v, acc_s, acc_v, cnt, N);
    ln_node_kernel<<<N, 128, 0, stream>>>(cur_s, cur_v, L.n1g, L.n1b, sn, vnb, 128, 16);
    gvp_node_kernel<160, 512><<<NB16, 256, 0, stream>>>(
        sn, vnb, N, 128, 16, 32, 512, 32,
        L.f0.wh, L.f0.ws_b, decW[l].f0, L.f0.wsv_b, decW[l].gf0, L.f0.wv, 1, fs, fv);
    gvp_node_kernel<544, 128><<<NB16, 256, 0, stream>>>(
        fs, fv, N, 512, 32, 32, 128, 16,
        L.f1.wh, L.f1.ws_b, decW[l].f1, L.f1.wsv_b, decW[l].gf1, L.f1.wv, 0, tmp_s, tmp_v);
    residual_add_kernel<<<(N * 176 + 255) / 256, 256, 0, stream>>>(cur_s, cur_v, tmp_s, tmp_v, N);
  }

  // ---- output head ---------------------------------------------------------
  wout_kernel<<<(N + 127) / 128, 128, 0, stream>>>(
      cur_s, cur_v, Wout_wh, Wout_b, Wout_w, (float*)d_out, N);

  (void)n_in; (void)out_size; (void)ws_size;
}
